// temporal_17712445129536
// MI455X (gfx1250) — matrix operand order repaired and checked by reading, not compiled, not run on hardware
//
#include <hip/hip_runtime.h>
#include <math.h>

typedef _Float16 h16;
typedef __attribute__((ext_vector_type(16))) _Float16 v16h;
typedef __attribute__((ext_vector_type(8)))  _Float16 v8h;
typedef __attribute__((ext_vector_type(4)))  _Float16 v4h;
typedef __attribute__((ext_vector_type(8)))  float    v8f;
typedef __attribute__((ext_vector_type(4)))  unsigned int u32x4;
typedef __attribute__((ext_vector_type(8)))  unsigned int u32x8;

#define DIMX 512
#define NTOK 1024
#define NBL  16
#define NH   8
#define DH   64
#define WELEMS (DIMX * DIMX)   // 262144 per weight matrix

// ---------------------------------------------------------------------------
// TDM: issue a 2D tile load Global->LDS (Tensor Data Mover, gfx1250).
// Descriptor per CDNA5 ISA 8.3/8.4: group0 (4 SGPRs) + group1 (8 SGPRs).
// data_size = 2 bytes (f16). tile_dim2 = 0 (2D). No padding, no multicast.
// ---------------------------------------------------------------------------
static __device__ __forceinline__ void tdm_load_2d(
    unsigned int lds_off, const h16* gptr,
    unsigned int tile_w, unsigned int tile_h,
    unsigned int tensor_w, unsigned int tensor_h,
    unsigned int stride_elems) {
  unsigned long long ga = (unsigned long long)(uintptr_t)gptr;
  u32x4 g0;
  g0[0] = 1u;                                   // count=1, user descriptor
  g0[1] = lds_off;                              // lds_addr (bytes)
  g0[2] = (unsigned int)ga;                     // global_addr[31:0]
  g0[3] = (unsigned int)((ga >> 32) & 0x01FFFFFFull) | (2u << 30);  // addr[56:32] | type=2
  u32x8 g1;
  g1[0] = 0x00010000u;                          // workgroup_mask=0, data_size=1 (2B)
  g1[1] = (tensor_w & 0xFFFFu) << 16;           // atomic_barrier=0, tensor_dim0[15:0]
  g1[2] = (tensor_w >> 16) | ((tensor_h & 0xFFFFu) << 16);
  g1[3] = (tensor_h >> 16) | ((tile_w & 0xFFFFu) << 16);  // tile_dim0
  g1[4] = tile_h & 0xFFFFu;                     // tile_dim1; tile_dim2=0
  g1[5] = stride_elems;                         // tensor_dim0_stride[31:0]
  g1[6] = 0u;
  g1[7] = 0u;
  asm volatile("tensor_load_to_lds %0, %1" :: "s"(g0), "s"(g1) : "memory");
}

static __device__ __forceinline__ unsigned int lds_off_of(const void* p) {
  return (unsigned int)(uintptr_t)p;  // flat LDS aperture: addr[31:0] = LDS offset
}

// ---------- fragment helpers (CDNA5 16x16x32 f16 WMMA VGPR layouts) ----------
static __device__ __forceinline__ v16h frag_a_f32(const float* rp, int kk, int hi) {
  const float4* p0 = reinterpret_cast<const float4*>(rp + kk + hi * 16);
  const float4* p1 = reinterpret_cast<const float4*>(rp + kk + hi * 16 + 8);
  float4 f0 = p0[0], f1 = p0[1], f2 = p1[0], f3 = p1[1];
  v16h a;
  a[0] = (h16)f0.x;  a[1] = (h16)f0.y;  a[2] = (h16)f0.z;  a[3] = (h16)f0.w;
  a[4] = (h16)f1.x;  a[5] = (h16)f1.y;  a[6] = (h16)f1.z;  a[7] = (h16)f1.w;
  a[8] = (h16)f2.x;  a[9] = (h16)f2.y;  a[10] = (h16)f2.z; a[11] = (h16)f2.w;
  a[12] = (h16)f3.x; a[13] = (h16)f3.y; a[14] = (h16)f3.z; a[15] = (h16)f3.w;
  return a;
}

static __device__ __forceinline__ v16h frag_a_f16(const h16* rp, int kk, int hi) {
  v8h lo = *reinterpret_cast<const v8h*>(rp + kk + hi * 16);
  v8h hh = *reinterpret_cast<const v8h*>(rp + kk + hi * 16 + 8);
  v16h a;
#pragma unroll
  for (int i = 0; i < 8; ++i) { a[i] = lo[i]; a[8 + i] = hh[i]; }
  return a;
}

static __device__ __forceinline__ v16h frag_b_f16(const h16* p) {
  return *reinterpret_cast<const v16h*>(p);
}

static __device__ __forceinline__ v8f vzero8() {
  v8f z = {0.f, 0.f, 0.f, 0.f, 0.f, 0.f, 0.f, 0.f};
  return z;
}

static __device__ __forceinline__ float redmax16(float v) {
#pragma unroll
  for (int m = 1; m < 16; m <<= 1) v = fmaxf(v, __shfl_xor(v, m, 32));
  return v;
}
static __device__ __forceinline__ float redsum16(float v) {
#pragma unroll
  for (int m = 1; m < 16; m <<= 1) v += __shfl_xor(v, m, 32);
  return v;
}

#define WMMA_F16(a, b, c) \
  __builtin_amdgcn_wmma_f32_16x16x32_f16(false, (a), false, (b), (short)0, (c), false, false)

// ---------------- kernel 0: convert 4 weight matrices fp32 -> f16 ------------
// Wh layout: [0]=Wq [1]=Wv [2]=Wk [3]=Wo, each 512x512 row-major.
__global__ __launch_bounds__(256) void cvtw_kernel(
    const float* __restrict__ Wq, const float* __restrict__ Wv,
    const float* __restrict__ Wk, const float* __restrict__ Wo,
    h16* __restrict__ dst) {
  const int idx = (blockIdx.x * 256 + threadIdx.x) * 4;
  const int mat = idx >> 18;
  const int loc = idx & (WELEMS - 1);
  const float* src = (mat == 0) ? Wq : (mat == 1) ? Wv : (mat == 2) ? Wk : Wo;
  float4 f = *reinterpret_cast<const float4*>(src + loc);
  v4h o;
  o[0] = (h16)f.x; o[1] = (h16)f.y; o[2] = (h16)f.z; o[3] = (h16)f.w;
  *reinterpret_cast<v4h*>(dst + idx) = o;
}

// ---------------- kernel 1: fused QKV projections ----------------------------
// mat 0: Q = x@Wq^T + bq (row-major f16)   mat 1: K = x@Wv^T + bv (bug-faithful)
// mat 2: V = x@Wk^T + bk (stored transposed [bl][512][1024])
// B tiles (64 cols x 32 k, shared by all 8 waves) staged via TDM, double-buffered.
// Epilogue: C-layout tile staged in wave-private LDS, stored as coalesced 32B.
__global__ __launch_bounds__(256) void proj_kernel(
    const float* __restrict__ x, const h16* __restrict__ Wh,
    const float* __restrict__ bq, const float* __restrict__ bv,
    const float* __restrict__ bk,
    h16* __restrict__ Qb, h16* __restrict__ Kb, h16* __restrict__ Vt) {
  __shared__ alignas(32) h16 ldsB[2][64 * 32];
  __shared__ alignas(32) h16 ldsT[8][16][68];  // padded vs bank conflicts
  const int lane = threadIdx.x & 31, w = threadIdx.x >> 5;
  const int hi = lane >> 4, ln = lane & 15;
  const int rb = blockIdx.x >> 3, cb = blockIdx.x & 7;
  const int bl = blockIdx.y, mat = blockIdx.z;

  const h16* Wm = Wh + (size_t)mat * WELEMS;
  const float* bias = (mat == 0) ? bq : (mat == 1) ? bv : bk;

  const int n0 = rb * 128 + w * 16;
  const int j0 = cb * 64;
  const float* xrow = x + (size_t)(bl * NTOK + n0 + ln) * DIMX;

  v8f acc[4];
#pragma unroll
  for (int nt = 0; nt < 4; ++nt) acc[nt] = vzero8();

  if (w == 0)
    tdm_load_2d(lds_off_of(&ldsB[0][0]), Wm + (size_t)j0 * DIMX,
                32, 64, DIMX, DIMX, DIMX);

  for (int it = 0; it < DIMX / 32; ++it) {
    const int kk = it * 32;
    __builtin_amdgcn_s_wait_tensorcnt(0);  // no-op for non-issuing waves
    __syncthreads();  // buffer it&1 ready; prior compute drained
    if (w == 0 && it + 1 < DIMX / 32)
      tdm_load_2d(lds_off_of(&ldsB[(it + 1) & 1][0]),
                  Wm + (size_t)j0 * DIMX + (kk + 32),
                  32, 64, DIMX, DIMX, DIMX);

    v16h a = frag_a_f32(xrow, kk, hi);
    const h16* bb = &ldsB[it & 1][0];
    v16h b0 = frag_b_f16(bb + (0 * 16 + ln) * 32 + hi * 16);
    v16h b1 = frag_b_f16(bb + (1 * 16 + ln) * 32 + hi * 16);
    v16h b2 = frag_b_f16(bb + (2 * 16 + ln) * 32 + hi * 16);
    v16h b3 = frag_b_f16(bb + (3 * 16 + ln) * 32 + hi * 16);
    acc[0] = WMMA_F16(a, b0, acc[0]);
    acc[1] = WMMA_F16(a, b1, acc[1]);
    acc[2] = WMMA_F16(a, b2, acc[2]);
    acc[3] = WMMA_F16(a, b3, acc[3]);
  }

  // epilogue: stage 16x64 tile (bias added) in wave-private LDS
#pragma unroll
  for (int nt = 0; nt < 4; ++nt) {
    const float bj = bias[j0 + nt * 16 + ln];
#pragma unroll
    for (int r = 0; r < 8; ++r)
      ldsT[w][r + 8 * hi][nt * 16 + ln] = (h16)(acc[nt][r] + bj);
  }
  if (mat < 2) {
    // row-major pack: each lane stores 32 contiguous halfs of one row
    h16* base = (mat == 0) ? Qb : Kb;
    const int row = lane >> 1, seg = (lane & 1) * 32;
    v16h d0, d1;
#pragma unroll
    for (int i = 0; i < 16; ++i) { d0[i] = ldsT[w][row][seg + i]; d1[i] = ldsT[w][row][seg + 16 + i]; }
    h16* dst = base + (size_t)(bl * NTOK + n0 + row) * DIMX + j0 + seg;
    *reinterpret_cast<v16h*>(dst) = d0;
    *reinterpret_cast<v16h*>(dst + 16) = d1;
  } else {
    // transpose pack: each lane gathers 2 columns, stores 32B rows of Vt
#pragma unroll
    for (int c = 0; c < 2; ++c) {
      const int jl = lane * 2 + c;
      v16h d;
#pragma unroll
      for (int i = 0; i < 16; ++i) d[i] = ldsT[w][i][jl];
      *reinterpret_cast<v16h*>(Vt + ((size_t)bl * DIMX + j0 + jl) * NTOK + n0) = d;
    }
  }
}

// ---------------- kernel 2: flash attention per (b,l,h) ----------------------
// K-chunk (32x64) and V^T-chunk (64x32) are identical across the 8 waves:
// both staged via TDM, double-buffered. P relays through wave-private LDS.
__global__ __launch_bounds__(256) void attn_kernel(
    const h16* Qb, const h16* Kb, const h16* Vt, h16* inter) {
  __shared__ alignas(32) h16 ldsK[2][32 * 64];
  __shared__ alignas(32) h16 ldsV[2][64 * 32];
  __shared__ alignas(32) h16 ldsP[8][16 * 32];
  __shared__ alignas(32) h16 ldsT[8][16][64];
  const int lane = threadIdx.x & 31, w = threadIdx.x >> 5;
  const int hi = lane >> 4, ln = lane & 15;
  const int q0 = blockIdx.x * 128 + w * 16;
  const int bl = blockIdx.y;
  const int hh = blockIdx.z;

  const h16* qrow = Qb + (size_t)(bl * NTOK + q0 + ln) * DIMX + hh * DH;
  v16h aq0 = frag_a_f16(qrow, 0, hi);
  v16h aq1 = frag_a_f16(qrow, 32, hi);

  v8f O[4];
#pragma unroll
  for (int nt = 0; nt < 4; ++nt) O[nt] = vzero8();
  float mrun[8], lrun[8];
#pragma unroll
  for (int r = 0; r < 8; ++r) { mrun[r] = -1e30f; lrun[r] = 0.f; }

  const h16* kbase = Kb + (size_t)(bl * NTOK) * DIMX + hh * DH;  // rows x 512
  const h16* vbase = Vt + ((size_t)bl * DIMX + hh * DH) * NTOK;  // rows x 1024
  const float scale = 0.044194173824159216f;  // 1/sqrt(512)

  if (w == 0) {
    tdm_load_2d(lds_off_of(&ldsK[0][0]), kbase, 64, 32, DIMX, NTOK, DIMX);
    tdm_load_2d(lds_off_of(&ldsV[0][0]), vbase, 32, 64, NTOK, DIMX, NTOK);
  }

  for (int it = 0; it < NTOK / 32; ++it) {
    const int m0 = it * 32;
    __builtin_amdgcn_s_wait_tensorcnt(0);  // no-op for non-issuing waves
    __syncthreads();
    if (w == 0 && it + 1 < NTOK / 32) {
      tdm_load_2d(lds_off_of(&ldsK[(it + 1) & 1][0]),
                  kbase + (size_t)(m0 + 32) * DIMX, 64, 32, DIMX, NTOK, DIMX);
      tdm_load_2d(lds_off_of(&ldsV[(it + 1) & 1][0]),
                  vbase + (m0 + 32), 32, 64, NTOK, DIMX, NTOK);
    }

    const h16* kb = &ldsK[it & 1][0];
    const h16* vb = &ldsV[it & 1][0];

    // S tiles: load all 4 B fragments, then run the 4 WMMAs back-to-back
    v16h bk0 = frag_b_f16(kb + ln * 64 + hi * 16);
    v16h bk1 = frag_b_f16(kb + ln * 64 + 32 + hi * 16);
    v16h bk2 = frag_b_f16(kb + (16 + ln) * 64 + hi * 16);
    v16h bk3 = frag_b_f16(kb + (16 + ln) * 64 + 32 + hi * 16);
    v8f s0 = vzero8(), s1 = vzero8();
    s0 = WMMA_F16(aq0, bk0, s0);
    s0 = WMMA_F16(aq1, bk1, s0);
    s1 = WMMA_F16(aq0, bk2, s1);
    s1 = WMMA_F16(aq1, bk3, s1);

    // online softmax: row M = r + 8*hi lives in this lane's half-wave
    float fac[8];
#pragma unroll
    for (int r = 0; r < 8; ++r) {
      float v0 = s0[r] * scale;
      float v1 = s1[r] * scale;
      float rm = fmaxf(redmax16(v0), redmax16(v1));
      float mnew = fmaxf(mrun[r], rm);
      float f = __expf(mrun[r] - mnew);
      float p0 = __expf(v0 - mnew);
      float p1 = __expf(v1 - mnew);
      lrun[r] = lrun[r] * f + redsum16(p0) + redsum16(p1);
      mrun[r] = mnew;
      fac[r] = f;
      ldsP[w][(r + 8 * hi) * 32 + ln] = (h16)p0;
      ldsP[w][(r + 8 * hi) * 32 + 16 + ln] = (h16)p1;
    }

#pragma unroll
    for (int nt = 0; nt < 4; ++nt)
#pragma unroll
      for (int r = 0; r < 8; ++r) O[nt][r] *= fac[r];

    // B fragments for P@V (independent of the P relay; issue early)
    v16h bv0 = frag_b_f16(vb + (0 * 16 + ln) * 32 + hi * 16);
    v16h bv1 = frag_b_f16(vb + (1 * 16 + ln) * 32 + hi * 16);
    v16h bv2 = frag_b_f16(vb + (2 * 16 + ln) * 32 + hi * 16);
    v16h bv3 = frag_b_f16(vb + (3 * 16 + ln) * 32 + hi * 16);

    // C/D-layout -> A-layout relay through wave-private LDS (in-order DS)
    v16h pa = frag_a_f16(&ldsP[w][ln * 32], 0, hi);

    O[0] = WMMA_F16(pa, bv0, O[0]);
    O[1] = WMMA_F16(pa, bv1, O[1]);
    O[2] = WMMA_F16(pa, bv2, O[2]);
    O[3] = WMMA_F16(pa, bv3, O[3]);
  }

  float linv[8];
#pragma unroll
  for (int r = 0; r < 8; ++r) linv[r] = 1.0f / lrun[r];

  // epilogue: stage normalized tile in wave-private LDS, store 64B per lane
#pragma unroll
  for (int nt = 0; nt < 4; ++nt)
#pragma unroll
    for (int r = 0; r < 8; ++r)
      ldsT[w][r + 8 * hi][nt * 16 + ln] = (h16)(O[nt][r] * linv[r]);
  {
    const int row = lane >> 1, seg = (lane & 1) * 32;
    v16h d0, d1;
#pragma unroll
    for (int i = 0; i < 16; ++i) { d0[i] = ldsT[w][row][seg + i]; d1[i] = ldsT[w][row][seg + 16 + i]; }
    h16* dst = inter + (size_t)(bl * NTOK + q0 + row) * DIMX + hh * DH + seg;
    *reinterpret_cast<v16h*>(dst) = d0;
    *reinterpret_cast<v16h*>(dst + 16) = d1;
  }
}

// ---------------- kernel 3: output projection --------------------------------
__global__ __launch_bounds__(256) void out_kernel(
    const h16* __restrict__ inter, const h16* __restrict__ Woh,
    const float* __restrict__ bo, float* __restrict__ out) {
  __shared__ alignas(32) h16 ldsB[2][64 * 32];
  const int lane = threadIdx.x & 31, w = threadIdx.x >> 5;
  const int hi = lane >> 4, ln = lane & 15;
  const int rb = blockIdx.x >> 3, cb = blockIdx.x & 7;
  const int bl = blockIdx.y;

  const int n0 = rb * 128 + w * 16;
  const int j0 = cb * 64;
  const h16* arow = inter + (size_t)(bl * NTOK + n0 + ln) * DIMX;

  v8f acc[4];
#pragma unroll
  for (int nt = 0; nt < 4; ++nt) acc[nt] = vzero8();

  if (w == 0)
    tdm_load_2d(lds_off_of(&ldsB[0][0]), Woh + (size_t)j0 * DIMX,
                32, 64, DIMX, DIMX, DIMX);

  for (int it = 0; it < DIMX / 32; ++it) {
    const int kk = it * 32;
    __builtin_amdgcn_s_wait_tensorcnt(0);
    __syncthreads();
    if (w == 0 && it + 1 < DIMX / 32)
      tdm_load_2d(lds_off_of(&ldsB[(it + 1) & 1][0]),
                  Woh + (size_t)j0 * DIMX + (kk + 32),
                  32, 64, DIMX, DIMX, DIMX);

    v16h a = frag_a_f16(arow, kk, hi);
    const h16* bb = &ldsB[it & 1][0];
    v16h b0 = frag_b_f16(bb + (0 * 16 + ln) * 32 + hi * 16);
    v16h b1 = frag_b_f16(bb + (1 * 16 + ln) * 32 + hi * 16);
    v16h b2 = frag_b_f16(bb + (2 * 16 + ln) * 32 + hi * 16);
    v16h b3 = frag_b_f16(bb + (3 * 16 + ln) * 32 + hi * 16);
    acc[0] = WMMA_F16(a, b0, acc[0]);
    acc[1] = WMMA_F16(a, b1, acc[1]);
    acc[2] = WMMA_F16(a, b2, acc[2]);
    acc[3] = WMMA_F16(a, b3, acc[3]);
  }

  // fp32 output: 16 lanes x b32 = 64B contiguous per (nt,r) — already coalesced
#pragma unroll
  for (int nt = 0; nt < 4; ++nt) {
    const int j = j0 + nt * 16 + ln;
    const float bj = bo[j];
#pragma unroll
    for (int r = 0; r < 8; ++r) {
      const int row = n0 + r + 8 * hi;
      out[(size_t)(bl * NTOK + row) * DIMX + j] = acc[nt][r] + bj;
    }
  }
}

// ---------------- launch ------------------------------------------------------
extern "C" void kernel_launch(void* const* d_in, const int* in_sizes, int n_in,
                              void* d_out, int out_size, void* d_ws, size_t ws_size,
                              hipStream_t stream) {
  (void)in_sizes; (void)n_in; (void)out_size; (void)ws_size;
  const float* x  = (const float*)d_in[0];
  const float* Wq = (const float*)d_in[1];
  const float* bq = (const float*)d_in[2];
  const float* Wk = (const float*)d_in[3];
  const float* bk = (const float*)d_in[4];
  const float* Wv = (const float*)d_in[5];
  const float* bv = (const float*)d_in[6];
  const float* Wo = (const float*)d_in[7];
  const float* bo = (const float*)d_in[8];

  const size_t BUF = (size_t)NBL * NTOK * DIMX;  // 8M f16 = 16 MB
  h16* Wh = (h16*)d_ws;               // 4 x 512 x 512 f16 = 2 MB  [Wq, Wv, Wk, Wo]
  h16* Qb = Wh + (size_t)4 * WELEMS;  // also reused as `inter` (safe aliasing)
  h16* Kb = Qb + BUF;
  h16* Vt = Kb + BUF;

  cvtw_kernel<<<dim3(4 * WELEMS / 1024), 256, 0, stream>>>(Wq, Wv, Wk, Wo, Wh);
  proj_kernel<<<dim3(64, NBL, 3), 256, 0, stream>>>(x, Wh, bq, bv, bk, Qb, Kb, Vt);
  attn_kernel<<<dim3(8, NBL, NH), 256, 0, stream>>>(Qb, Kb, Vt, Qb);
  out_kernel<<<dim3(64, NBL), 256, 0, stream>>>(Qb, Wh + (size_t)3 * WELEMS, bo,
                                                (float*)d_out);
}
